// CLIPAttention_39986145525806
// MI455X (gfx1250) — matrix operand order repaired and checked by reading, not compiled, not run on hardware
//
#include <hip/hip_runtime.h>
#include <hip/hip_bf16.h>

// Problem constants (from reference setup_inputs)
#define B_   16
#define M_   4
#define N_   12
#define L_   196
#define E_   768
#define H_   12
#define D_   64
#define T_   2356          // M + N*L
#define BT_  37696         // B*T
#define KTOT_ 200          // M + L keys per frame
#define KPAD_ 224          // padded key dim (multiple of 32)

typedef __attribute__((ext_vector_type(16))) _Float16 v16h;
typedef __attribute__((ext_vector_type(8)))  _Float16 v8h;
typedef __attribute__((ext_vector_type(8)))  float    v8f;
typedef __attribute__((ext_vector_type(4)))  float    v4f;

static __device__ inline v16h concat8(v8h lo, v8h hi) {
  return __builtin_shufflevector(lo, hi, 0,1,2,3,4,5,6,7,8,9,10,11,12,13,14,15);
}

// A-fragment (16x32, f16) per ISA layout:
// lanes 0-15 : row = lane,     K in {k0+0..7, k0+16..23}
// lanes 16-31: row = lane-16,  K in {k0+8..15, k0+24..31}
static __device__ inline v16h load_a_f16(const _Float16* rowptr, int k0, int lh) {
  v8h lo = *(const v8h*)(rowptr + k0 + lh * 8);
  v8h hi = *(const v8h*)(rowptr + k0 + 16 + lh * 8);
  return concat8(lo, hi);
}

static __device__ inline v16h load_a_f32(const float* rowptr, int k0, int lh) {
  v4f f0 = *(const v4f*)(rowptr + k0 + lh * 8);
  v4f f1 = *(const v4f*)(rowptr + k0 + lh * 8 + 4);
  v4f f2 = *(const v4f*)(rowptr + k0 + 16 + lh * 8);
  v4f f3 = *(const v4f*)(rowptr + k0 + 16 + lh * 8 + 4);
  v16h a;
#pragma unroll
  for (int i = 0; i < 4; ++i) {
    a[i]      = (_Float16)f0[i];
    a[4 + i]  = (_Float16)f1[i];
    a[8 + i]  = (_Float16)f2[i];
    a[12 + i] = (_Float16)f3[i];
  }
  return a;
}

// B-fragment (32x16, f16) from a [N][K] row-major (N-major) buffer:
// lane holds column `col`, K in {k0+lh*8..+7, k0+16+lh*8..+7} (matches A layout)
static __device__ inline v16h load_b_f16(const _Float16* base, int stride, int col, int k0, int lh) {
  const _Float16* p = base + (size_t)col * stride + k0 + lh * 8;
  return concat8(*(const v8h*)p, *(const v8h*)(p + 16));
}

static __device__ inline v8f wmma_f16(v16h a, v16h b, v8f c) {
  return __builtin_amdgcn_wmma_f32_16x16x32_f16(false, a, false, b, (short)0, c, false, false);
}

// ---------------------------------------------------------------------------
// Weight prep: WT[n*768 + k] = (f16) W[k*768 + n]  for 4 matrices
// ---------------------------------------------------------------------------
__global__ __launch_bounds__(256) void prep_w(
    const float* W0, const float* W1, const float* W2, const float* W3,
    _Float16* T0, _Float16* T1, _Float16* T2, _Float16* T3) {
  const float* Ws[4] = {W0, W1, W2, W3};
  _Float16*    Ts[4] = {T0, T1, T2, T3};
  int mat = blockIdx.y;
  int e = blockIdx.x * 256 + threadIdx.x;   // 0 .. 589823
  int nn = e / E_;
  int kk = e - nn * E_;
  Ts[mat][(size_t)nn * E_ + kk] = (_Float16)Ws[mat][(size_t)kk * E_ + nn];
}

// ---------------------------------------------------------------------------
// GEMM: Out[row][col] = (A[row][:] . WT[col][:] + bias[col]) * scale
// One wave computes one 16x64 tile: one A-fragment feeds 4 WMMAs per k-step
// (24 k-steps, 96 WMMAs per wave).
// A_F32: A is fp32 (hidden states), output f16.  Else: A f16, output fp32.
// ---------------------------------------------------------------------------
template <bool A_F32>
__global__ __launch_bounds__(256) void gemm16x64(
    const void* __restrict__ Aptr, const _Float16* __restrict__ WT,
    const float* __restrict__ bias, void* __restrict__ Out, float scale) {
  int tid  = threadIdx.x;
  int wave = tid >> 5, lane = tid & 31, lh = lane >> 4, ln = lane & 15;
  int tile = blockIdx.x * 8 + wave;          // (rowTile, colGroup) pairs
  int rowTile = tile / (E_ / 64);            // 12 col groups of 64
  int cg      = tile - rowTile * (E_ / 64);
  int row = rowTile * 16 + ln;
  int n0  = cg * 64;

  v8f acc[4] = {};
  for (int k0 = 0; k0 < E_; k0 += 32) {
    v16h a;
    if (A_F32) a = load_a_f32((const float*)Aptr + (size_t)row * E_, k0, lh);
    else       a = load_a_f16((const _Float16*)Aptr + (size_t)row * E_, k0, lh);
#pragma unroll
    for (int nt = 0; nt < 4; ++nt) {
      v16h b = load_b_f16(WT, E_, n0 + nt * 16 + ln, k0, lh);
      acc[nt] = wmma_f16(a, b, acc[nt]);
    }
  }

#pragma unroll
  for (int nt = 0; nt < 4; ++nt) {
    int col = n0 + nt * 16 + ln;
    float bv = bias[col];
#pragma unroll
    for (int g = 0; g < 8; ++g) {
      int r = rowTile * 16 + g + lh * 8;   // C layout: lanes 0-15 rows 0-7, 16-31 rows 8-15
      float v = (acc[nt][g] + bv) * scale;
      if (A_F32) ((_Float16*)Out)[(size_t)r * E_ + col] = (_Float16)v;
      else       ((float*)Out)[(size_t)r * E_ + col] = v;
    }
  }
}

// ---------------------------------------------------------------------------
// Frame attention: one workgroup per (b, h, frame). 8 waves, each wave owns a
// 16-query tile (loop over 13 tiles).  S = Q K^T via WMMA, softmax in f32 with
// width-16 shuffle reductions, P staged in LDS as f16, O = P V via WMMA.
// K staged via CDNA5 async global->LDS DMA (ASYNCcnt).
// ---------------------------------------------------------------------------
__global__ __launch_bounds__(256) void frame_attn(
    const _Float16* __restrict__ Qh, const _Float16* __restrict__ Kh,
    const _Float16* __restrict__ Vh, _Float16* __restrict__ AOh) {
  extern __shared__ _Float16 smem[];
  _Float16* Ks = smem;                    // [208][64]   row-major keys (LDS offset 0)
  _Float16* Vt = Ks + 208 * 64;           // [64][KPAD_] transposed values
  _Float16* Pb = Vt + 64 * KPAD_;         // [8][16][KPAD_] probabilities

  int blk = blockIdx.x;
  int b = blk / (H_ * N_);
  int rem = blk - b * (H_ * N_);
  int h = rem / N_;
  int n = rem - h * N_;
  size_t fbase = (size_t)b * T_ + M_ + (size_t)n * L_;   // first frame token row

  int tid = threadIdx.x;

  // ---- zero K padding rows 200..207 (regular DS stores) ----
  for (int e = tid; e < 8 * 64; e += 256) Ks[KTOT_ * 64 + e] = (_Float16)0.f;

  // ---- stage K rows 0..199 via async global->LDS (16B per lane-chunk) ----
  // dynamic LDS starts at offset 0 in this kernel, so byte offset == LDS addr
  for (int idx = tid; idx < KTOT_ * 8; idx += 256) {
    int j = idx >> 3, c = idx & 7;
    size_t tok = (size_t)b * T_ + (j < M_ ? j : M_ + n * L_ + (j - M_));
    const _Float16* gptr = Kh + tok * E_ + h * D_ + c * 8;
    unsigned int ldsoff = (unsigned int)(j * 64 + c * 8) * 2u;   // bytes
    asm volatile("global_load_async_to_lds_b128 %0, %1, off"
                 :: "v"(ldsoff), "v"(gptr) : "memory");
  }

  // ---- stage V transposed (zero-padded cols) ----
  for (int e = tid; e < 64 * KPAD_; e += 256) {
    int j = e >> 6, dd = e & 63;
    _Float16 v = (_Float16)0.f;
    if (j < KTOT_) {
      size_t tok = (size_t)b * T_ + (j < M_ ? j : M_ + n * L_ + (j - M_));
      v = Vh[tok * E_ + h * D_ + dd];
    }
    Vt[dd * KPAD_ + j] = v;
  }
  // ---- zero P staging area ----
  for (int e = tid; e < 8 * 16 * KPAD_; e += 256) Pb[e] = (_Float16)0.f;

  asm volatile("s_wait_asynccnt 0x0" ::: "memory");   // K DMA complete
  __syncthreads();

  int wave = tid >> 5, lane = tid & 31, lh = lane >> 4, ln = lane & 15;
  _Float16* Pw = Pb + wave * 16 * KPAD_;

  for (int qt = wave; qt < 13; qt += 8) {
    int qrow = qt * 16 + ln;
    int qr = qrow < L_ ? qrow : (L_ - 1);      // clamp; clamped rows not stored
    const _Float16* qptr = Qh + (fbase + qr) * E_ + h * D_;
    v16h a0 = load_a_f16(qptr, 0, lh);
    v16h a1 = load_a_f16(qptr, 32, lh);

    // S = Q K^T : 13 tiles of 16 keys, d = 64 = 2 x 32
    v8f c[13] = {};
#pragma unroll
    for (int kt = 0; kt < 13; ++kt) {
      int j = kt * 16 + ln;
      v16h b0 = load_b_f16(Ks, 64, j, 0, lh);
      v16h b1 = load_b_f16(Ks, 64, j, 32, lh);
      c[kt] = wmma_f16(a0, b0, c[kt]);
      c[kt] = wmma_f16(a1, b1, c[kt]);
    }

    // softmax over 208 (masked to 200) keys, row r = g + lh*8
#pragma unroll
    for (int g = 0; g < 8; ++g) {
      float vals[13];
      float m = -1e30f;
#pragma unroll
      for (int kt = 0; kt < 13; ++kt) {
        float v = c[kt][g];
        if (kt * 16 + ln >= KTOT_) v = -1e30f;
        vals[kt] = v;
        m = fmaxf(m, v);
      }
#pragma unroll
      for (int o = 1; o < 16; o <<= 1) m = fmaxf(m, __shfl_xor(m, o, 16));
      float s = 0.f;
#pragma unroll
      for (int kt = 0; kt < 13; ++kt) {
        float p = (kt * 16 + ln < KTOT_) ? __expf(vals[kt] - m) : 0.f;
        vals[kt] = p;
        s += p;
      }
#pragma unroll
      for (int o = 1; o < 16; o <<= 1) s += __shfl_xor(s, o, 16);
      float inv = 1.f / s;
      int r = g + lh * 8;
#pragma unroll
      for (int kt = 0; kt < 13; ++kt)
        Pw[r * KPAD_ + kt * 16 + ln] = (_Float16)(vals[kt] * inv);
    }
    asm volatile("s_wait_dscnt 0x0" ::: "memory");   // LDS RAW fence (same wave)

    // O = P V : K-dim = 224 keys (7 x 32), N-dim = 64 (4 x 16)
    v8f o[4] = {};
#pragma unroll
    for (int kt2 = 0; kt2 < 7; ++kt2) {
      int k0 = kt2 * 32;
      v16h a = load_a_f16(Pw + (size_t)ln * KPAD_, k0, lh);
#pragma unroll
      for (int nt = 0; nt < 4; ++nt) {
        v16h bf = load_b_f16(Vt, KPAD_, nt * 16 + ln, k0, lh);
        o[nt] = wmma_f16(a, bf, o[nt]);
      }
    }

#pragma unroll
    for (int nt = 0; nt < 4; ++nt)
#pragma unroll
      for (int g = 0; g < 8; ++g) {
        int q2 = qt * 16 + g + lh * 8;
        if (q2 < L_)
          AOh[(fbase + q2) * E_ + h * D_ + nt * 16 + ln] = (_Float16)o[nt][g];
      }
  }
}

// ---------------------------------------------------------------------------
// CLS attention: one workgroup per (b, h); M=4 queries attend to all T keys.
// Tiny FLOPs -> plain VALU.
// ---------------------------------------------------------------------------
__global__ __launch_bounds__(256) void cls_attn(
    const _Float16* __restrict__ Qh, const _Float16* __restrict__ Kh,
    const _Float16* __restrict__ Vh, _Float16* __restrict__ AOh) {
  __shared__ float sl[M_ * T_];     // logits / probs
  __shared__ float qs[M_ * D_];
  __shared__ float red[M_ * 8];

  int blk = blockIdx.x;
  int b = blk / H_, h = blk - b * H_;
  int tid = threadIdx.x, wave = tid >> 5;
  size_t base = (size_t)b * T_;

  {
    int m = tid >> 6, dd = tid & 63;
    qs[tid] = (float)Qh[(base + m) * E_ + h * D_ + dd];   // pre-scaled Q
  }
  __syncthreads();

  float lmax[M_] = {-1e30f, -1e30f, -1e30f, -1e30f};
  for (int j = tid; j < T_; j += 256) {
    const _Float16* kp = Kh + (base + j) * E_ + h * D_;
    float s0 = 0.f, s1 = 0.f, s2 = 0.f, s3 = 0.f;
#pragma unroll 8
    for (int dd = 0; dd < D_; ++dd) {
      float kv = (float)kp[dd];
      s0 += qs[0 * D_ + dd] * kv;
      s1 += qs[1 * D_ + dd] * kv;
      s2 += qs[2 * D_ + dd] * kv;
      s3 += qs[3 * D_ + dd] * kv;
    }
    sl[0 * T_ + j] = s0; sl[1 * T_ + j] = s1;
    sl[2 * T_ + j] = s2; sl[3 * T_ + j] = s3;
    lmax[0] = fmaxf(lmax[0], s0); lmax[1] = fmaxf(lmax[1], s1);
    lmax[2] = fmaxf(lmax[2], s2); lmax[3] = fmaxf(lmax[3], s3);
  }
#pragma unroll
  for (int m = 0; m < M_; ++m) {
#pragma unroll
    for (int o = 16; o >= 1; o >>= 1) lmax[m] = fmaxf(lmax[m], __shfl_xor(lmax[m], o, 32));
    if ((tid & 31) == 0) red[m * 8 + wave] = lmax[m];
  }
  __syncthreads();
  float gmax[M_];
#pragma unroll
  for (int m = 0; m < M_; ++m) {
    float g = red[m * 8];
#pragma unroll
    for (int w = 1; w < 8; ++w) g = fmaxf(g, red[m * 8 + w]);
    gmax[m] = g;
  }
  __syncthreads();

  float lsum[M_] = {0.f, 0.f, 0.f, 0.f};
  for (int j = tid; j < T_; j += 256) {
#pragma unroll
    for (int m = 0; m < M_; ++m) {
      float p = __expf(sl[m * T_ + j] - gmax[m]);
      sl[m * T_ + j] = p;
      lsum[m] += p;
    }
  }
#pragma unroll
  for (int m = 0; m < M_; ++m) {
#pragma unroll
    for (int o = 16; o >= 1; o >>= 1) lsum[m] += __shfl_xor(lsum[m], o, 32);
    if ((tid & 31) == 0) red[m * 8 + wave] = lsum[m];
  }
  __syncthreads();
  float inv[M_];
#pragma unroll
  for (int m = 0; m < M_; ++m) {
    float s = 0.f;
#pragma unroll
    for (int w = 0; w < 8; ++w) s += red[m * 8 + w];
    inv[m] = 1.f / s;
  }
  __syncthreads();

  // thread <-> (m, dd) output element
  {
    int m = tid >> 6, dd = tid & 63;
    float acc = 0.f;
    for (int j = 0; j < T_; ++j)
      acc += sl[m * T_ + j] * (float)Vh[(base + j) * E_ + h * D_ + dd];
    AOh[(base + m) * E_ + h * D_ + dd] = (_Float16)(acc * inv[m]);
  }
}

// ---------------------------------------------------------------------------
extern "C" void kernel_launch(void* const* d_in, const int* in_sizes, int n_in,
                              void* d_out, int out_size, void* d_ws, size_t ws_size,
                              hipStream_t stream) {
  (void)in_sizes; (void)n_in; (void)out_size; (void)ws_size;
  const float* hs = (const float*)d_in[0];
  const float* Wq = (const float*)d_in[1];
  const float* bq = (const float*)d_in[2];
  const float* Wk = (const float*)d_in[3];
  const float* bk = (const float*)d_in[4];
  const float* Wv = (const float*)d_in[5];
  const float* bv = (const float*)d_in[6];
  const float* Wo = (const float*)d_in[7];
  const float* bo = (const float*)d_in[8];
  float* out = (float*)d_out;

  const size_t WSZ = (size_t)E_ * E_;        // 589824 halves per weight
  const size_t ASZ = (size_t)BT_ * E_;       // 28,950,528 halves per activation
  _Float16* ws  = (_Float16*)d_ws;
  _Float16* WqT = ws;
  _Float16* WkT = WqT + WSZ;
  _Float16* WvT = WkT + WSZ;
  _Float16* WoT = WvT + WSZ;
  _Float16* Qh  = WoT + WSZ;
  _Float16* Kh  = Qh + ASZ;
  _Float16* Vh  = Kh + ASZ;
  _Float16* AOh = Vh + ASZ;

  // 1) weight transpose + f16 convert
  prep_w<<<dim3((E_ * E_) / 256, 4), 256, 0, stream>>>(Wq, Wk, Wv, Wo, WqT, WkT, WvT, WoT);

  // 2) QKV projections (fp32 A -> f16 out); Q folded with d^-0.5 = 0.125
  const int gemm_blocks = (BT_ / 16) * (E_ / 64) / 8;   // 3534
  gemm16x64<true><<<gemm_blocks, 256, 0, stream>>>(hs, WqT, bq, Qh, 0.125f);
  gemm16x64<true><<<gemm_blocks, 256, 0, stream>>>(hs, WkT, bk, Kh, 1.0f);
  gemm16x64<true><<<gemm_blocks, 256, 0, stream>>>(hs, WvT, bv, Vh, 1.0f);

  // 3) attention
  const size_t frame_lds = (208 * 64 + 64 * KPAD_ + 8 * 16 * KPAD_) * sizeof(_Float16);
  frame_attn<<<B_ * H_ * N_, 256, frame_lds, stream>>>(Qh, Kh, Vh, AOh);
  cls_attn<<<B_ * H_, 256, 0, stream>>>(Qh, Kh, Vh, AOh);

  // 4) output projection (f16 A -> fp32 out)
  gemm16x64<false><<<gemm_blocks, 256, 0, stream>>>(AOh, WoT, bo, out, 1.0f);
}
